// PointNet2MSG_73521250173249
// MI455X (gfx1250) — matrix operand order repaired and checked by reading, not compiled, not run on hardware
//
#include <hip/hip_runtime.h>

// ---------------------------------------------------------------------------
// PointNet++ MSG forward for MI455X (gfx1250).
// All matrix work goes through a bf16 WMMA GEMM (v_wmma_f32_16x16x32_bf16)
// with LDS tiles stored as packed bf16x2 K-pairs so each fragment is eight
// dword ds_load_b32 reads (no half-register merging).
// d_in layout assumption: d_in[0] = points (B,N,4); remaining inputs are the
// leaves of params in JAX pytree order: dict keys sorted -> 'fp' then 'sa',
// each layer dict sorted -> bias, scale, w.
// ---------------------------------------------------------------------------

#define BATCH 4
#define NPTS  16384

typedef __bf16 bf16_t;
typedef bf16_t v16bf __attribute__((ext_vector_type(16)));
typedef float  v8f   __attribute__((ext_vector_type(8)));
typedef unsigned int v8u __attribute__((ext_vector_type(8)));

union frag_cast { v8u u; v16bf b; };
union pack_cast { bf16_t h[2]; unsigned int u; };

__device__ __forceinline__ unsigned int pack_bf16x2(float lo, float hi) {
    pack_cast p;
    p.h[0] = (bf16_t)lo;
    p.h[1] = (bf16_t)hi;
    return p.u;
}

// ============================ WMMA GEMM ====================================
// out = relu((A @ W) * scale + bias), A: MxK f32 row-major, W: KxN f32.
// Block tile 64(M) x 128(N), K-step 32, 8 waves arranged 4(M) x 2(N);
// each wave computes a 16x64 strip as 4 WMMA 16x16 tiles.
// Fused-A mode (idx3 != nullptr): A[m][k] = (k < Kinterp)
//     ? sum_j w3[m][j] * knownF[idx3[m][j]][k]           (3-NN interpolation)
//     : skipF[m*skipStride + (k - Kinterp)]              (skip-connection)
#define TM 64
#define TN 128
#define TKS 32
#define TKP (TKS / 2)   // packed K-pairs per tile

__global__ void __launch_bounds__(256)
wmma_gemm_bias_relu(const float* __restrict__ A,
                    const float* __restrict__ W,
                    const float* __restrict__ scale,
                    const float* __restrict__ bias,
                    float* __restrict__ out,
                    int M, int K, int N,
                    const int*   __restrict__ idx3,
                    const float* __restrict__ w3,
                    const float* __restrict__ knownF,
                    int Kinterp,
                    const float* __restrict__ skipF,
                    int skipStride)
{
    // LDS tiles: 32-bit elements each holding bf16(k), bf16(k+1).
    __shared__ unsigned int Aspk[TM][TKP + 2];   // 64 x 18 dw = 4.5 KB
    __shared__ unsigned int Bspk[TKP][TN + 4];   // 16 x 132 dw = 8.25 KB

    const int tid  = threadIdx.x;
    const int lane = tid & 31;
    const int wid  = tid >> 5;
    const int wm   = wid & 3;   // wave M group: 16 rows each
    const int wn   = wid >> 2;  // wave N group: 64 cols each
    const int m0   = blockIdx.y * TM;
    const int n0   = blockIdx.x * TN;

    v8f acc[4] = {};  // four 16x16 f32 accumulator tiles along N

    for (int k0 = 0; k0 < K; k0 += TKS) {
        // ---- stage A tile (TM x TKS) packed as bf16x2, zero padded ----
        for (int e = tid; e < TM * TKP; e += 256) {
            int r = e >> 4, p = e & (TKP - 1);
            int m = m0 + r;
            float v0 = 0.f, v1 = 0.f;
            if (m < M) {
                int k = k0 + 2 * p;
                if (idx3) {
                    const int*   id = idx3 + (size_t)m * 3;
                    const float* w  = w3   + (size_t)m * 3;
#pragma unroll
                    for (int q = 0; q < 2; ++q) {
                        int kk = k + q;
                        float v = 0.f;
                        if (kk < Kinterp) {
                            v = w[0] * knownF[(size_t)id[0] * Kinterp + kk]
                              + w[1] * knownF[(size_t)id[1] * Kinterp + kk]
                              + w[2] * knownF[(size_t)id[2] * Kinterp + kk];
                        } else if (kk < K) {
                            v = skipF[(size_t)m * skipStride + (kk - Kinterp)];
                        }
                        if (q == 0) v0 = v; else v1 = v;
                    }
                } else {
                    const float* ar = A + (size_t)m * K;
                    if (k < K)     v0 = ar[k];
                    if (k + 1 < K) v1 = ar[k + 1];
                }
            }
            Aspk[r][p] = pack_bf16x2(v0, v1);  // v_cvt_pk_bf16_f32 path
        }
        // ---- stage B tile (TKS x TN) packed as bf16x2 along K ----
        for (int e = tid; e < TKP * TN; e += 256) {
            int p = e >> 7, c = e & 127;
            int k = k0 + 2 * p, n = n0 + c;
            float v0 = 0.f, v1 = 0.f;
            if (n < N) {
                if (k < K)     v0 = W[(size_t)k * N + n];
                if (k + 1 < K) v1 = W[(size_t)(k + 1) * N + n];
            }
            Bspk[p][c] = pack_bf16x2(v0, v1);
        }
        __syncthreads();

        // prefetch next A K-slab (emits global_prefetch_b8 on gfx1250)
        if (A && (k0 + TKS) < K) {
            int pm = m0 + (tid & 63);
            if (pm < M)
                __builtin_prefetch(A + (size_t)pm * K + (k0 + TKS), 0, 1);
        }

        // ---- A fragment: 8 dword LDS reads (CDNA5 16-bit A layout) ----
        // lane = m + 16*half; VGPR v holds K-pair (v + 4*half) for v<4,
        // (8 + v-4 + 4*half) for v>=4.
        frag_cast fa;
        {
            const int am   = wm * 16 + (lane & 15);
            const int half = lane >> 4;
#pragma unroll
            for (int v = 0; v < 8; ++v) {
                int p = (v < 4) ? (v + 4 * half) : (8 + (v - 4) + 4 * half);
                fa.u[v] = Aspk[am][p];
            }
        }
        // ---- four B fragments + WMMA ----
        // lane = n + 16*half; VGPR v holds K-pair (v + 4*half) for v<4,
        // (8 + v-4 + 4*half) for v>=4.
#pragma unroll
        for (int t = 0; t < 4; ++t) {
            frag_cast fb;
            const int bn   = wn * 64 + t * 16 + (lane & 15);
            const int half = lane >> 4;
#pragma unroll
            for (int v = 0; v < 8; ++v)
                fb.u[v] = Bspk[(v < 4) ? (v + 4 * half) : (8 + (v - 4) + 4 * half)][bn];
            acc[t] = __builtin_amdgcn_wmma_f32_16x16x32_bf16(
                false, fa.b, false, fb.b, (short)0, acc[t], false, false);
        }
        __syncthreads();
    }

    // ---- epilogue: scale/bias/relu. C/D layout: lane = n + 16*(m/8) ----
#pragma unroll
    for (int t = 0; t < 4; ++t) {
        int n = n0 + wn * 64 + t * 16 + (lane & 15);
        if (n < N) {
            float sc = scale[n], bi = bias[n];
#pragma unroll
            for (int r = 0; r < 8; ++r) {
                int m = m0 + wm * 16 + r + 8 * (lane >> 4);
                if (m < M) {
                    float y = acc[t][r] * sc + bi;
                    out[(size_t)m * N + n] = y > 0.f ? y : 0.f;
                }
            }
        }
    }
}

// ======================= Farthest point sampling ===========================
// One block per batch. Matches jnp scan: out[0]=0; out[i]=argmax after update.
__global__ void __launch_bounds__(256)
fps_kernel(const float* __restrict__ xyz, int stride, int n, int npoint,
           float* __restrict__ dist, int* __restrict__ out)
{
    const int b = blockIdx.x;
    const int t = threadIdx.x;
    const int T = blockDim.x;
    const float* x = xyz + (size_t)b * n * stride;
    float* d = dist + (size_t)b * n;
    for (int j = t; j < n; j += T) d[j] = 1e10f;
    __syncthreads();

    __shared__ float sv[256];
    __shared__ int   si[256];
    int last = 0;
    for (int i = 0; i < npoint; ++i) {
        if (t == 0) out[b * npoint + i] = last;
        float lx = x[(size_t)last * stride];
        float ly = x[(size_t)last * stride + 1];
        float lz = x[(size_t)last * stride + 2];
        float bestv = -1.f; int besti = 0;
        for (int j = t; j < n; j += T) {
            float dx = x[(size_t)j * stride]     - lx;
            float dy = x[(size_t)j * stride + 1] - ly;
            float dz = x[(size_t)j * stride + 2] - lz;
            float nd = dx * dx + dy * dy + dz * dz;
            float nn = fminf(d[j], nd);
            d[j] = nn;
            if (nn > bestv) { bestv = nn; besti = j; }
        }
        sv[t] = bestv; si[t] = besti;
        __syncthreads();
        for (int s = T / 2; s > 0; s >>= 1) {
            if (t < s) {
                if (sv[t + s] > sv[t] ||
                    (sv[t + s] == sv[t] && si[t + s] < si[t])) {
                    sv[t] = sv[t + s]; si[t] = si[t + s];
                }
            }
            __syncthreads();
        }
        last = si[0];
        __syncthreads();
    }
}

__global__ void gather_xyz_kernel(const float* __restrict__ xyz, int stride,
                                  const int* __restrict__ idx,
                                  int B_, int n, int S,
                                  float* __restrict__ out)
{
    size_t i = blockIdx.x * (size_t)blockDim.x + threadIdx.x;
    if (i >= (size_t)B_ * S) return;
    int b = (int)(i / S);
    int j = idx[i];
    const float* p = xyz + ((size_t)b * n + j) * stride;
    out[i * 3 + 0] = p[0];
    out[i * 3 + 1] = p[1];
    out[i * 3 + 2] = p[2];
}

// ====================== Ball query + grouping ==============================
// One thread per (b,s): first `nsample` points with d2<r2 in index order,
// padded with the first match. Writes rows [dx,dy,dz, feat...].
__global__ void ball_group_kernel(const float* __restrict__ xyz, int xstride,
                                  const float* __restrict__ feat, int featC, int fstride,
                                  const float* __restrict__ centers,
                                  int B_, int n, int S, int nsample, float r2,
                                  float* __restrict__ grp)
{
    size_t i = blockIdx.x * (size_t)blockDim.x + threadIdx.x;
    if (i >= (size_t)B_ * S) return;
    int b = (int)(i / S);
    const float* x = xyz + (size_t)b * n * xstride;
    const float* f = feat + (size_t)b * n * fstride;
    const float* c = centers + i * 3;
    const float cx = c[0], cy = c[1], cz = c[2];
    const int C = 3 + featC;
    float* g = grp + i * (size_t)nsample * C;

    int cnt = 0;
    for (int j = 0; j < n && cnt < nsample; ++j) {
        float dx = x[(size_t)j * xstride]     - cx;
        float dy = x[(size_t)j * xstride + 1] - cy;
        float dz = x[(size_t)j * xstride + 2] - cz;
        float d = dx * dx + dy * dy + dz * dz;
        if (d < r2) {
            float* row = g + (size_t)cnt * C;
            row[0] = dx; row[1] = dy; row[2] = dz;
            for (int cc = 0; cc < featC; ++cc)
                row[3 + cc] = f[(size_t)j * fstride + cc];
            ++cnt;
        }
    }
    if (cnt == 0) {  // degenerate fallback: point 0
        float dx = x[0] - cx, dy = x[1] - cy, dz = x[2] - cz;
        g[0] = dx; g[1] = dy; g[2] = dz;
        for (int cc = 0; cc < featC; ++cc) g[3 + cc] = f[cc];
        cnt = 1;
    }
    for (int k = cnt; k < nsample; ++k) {
        float* row = g + (size_t)k * C;
        for (int cc = 0; cc < C; ++cc) row[cc] = g[cc];
    }
}

// ============================ Max pool =====================================
__global__ void maxpool_kernel(const float* __restrict__ in, int ns, int C,
                               float* __restrict__ out, int Ctot, int colOff,
                               size_t total)
{
    size_t i = blockIdx.x * (size_t)blockDim.x + threadIdx.x;
    if (i >= total) return;
    size_t s = i / C;
    int c = (int)(i % C);
    const float* p = in + s * (size_t)ns * C + c;
    float m = p[0];
    for (int j = 1; j < ns; ++j) m = fmaxf(m, p[(size_t)j * C]);
    out[s * (size_t)Ctot + colOff + c] = m;
}

// ========================= Three nearest neighbors =========================
__global__ void three_nn_kernel(const float* __restrict__ unknown, int ustride,
                                const float* __restrict__ known, int kstride,
                                int B_, int Nu, int S,
                                int* __restrict__ idx3, float* __restrict__ w3)
{
    size_t i = blockIdx.x * (size_t)blockDim.x + threadIdx.x;
    if (i >= (size_t)B_ * Nu) return;
    int b = (int)(i / Nu);
    const float* u  = unknown + i * (size_t)ustride;
    const float* kb = known + (size_t)b * S * kstride;
    float ux = u[0], uy = u[1], uz = u[2];
    float d0 = 3.4e38f, d1 = 3.4e38f, d2v = 3.4e38f;
    int i0 = 0, i1 = 0, i2 = 0;
    for (int j = 0; j < S; ++j) {
        float dx = kb[(size_t)j * kstride]     - ux;
        float dy = kb[(size_t)j * kstride + 1] - uy;
        float dz = kb[(size_t)j * kstride + 2] - uz;
        float d = dx * dx + dy * dy + dz * dz;
        if (d < d0)       { d2v = d1; i2 = i1; d1 = d0; i1 = i0; d0 = d; i0 = j; }
        else if (d < d1)  { d2v = d1; i2 = i1; d1 = d;  i1 = j; }
        else if (d < d2v) { d2v = d;  i2 = j; }
    }
    float w0 = 1.f / (d0 + 1e-8f);
    float w1 = 1.f / (d1 + 1e-8f);
    float w2 = 1.f / (d2v + 1e-8f);
    float s = w0 + w1 + w2;
    idx3[i * 3 + 0] = b * S + i0;  // absolute row into known feature matrix
    idx3[i * 3 + 1] = b * S + i1;
    idx3[i * 3 + 2] = b * S + i2;
    w3[i * 3 + 0] = w0 / s;
    w3[i * 3 + 1] = w1 / s;
    w3[i * 3 + 2] = w2 / s;
}

// ============================ Transpose ====================================
// (B, Npts, C) -> (B, C, Npts)
__global__ void transpose_kernel(const float* __restrict__ in,
                                 float* __restrict__ out,
                                 int B_, int Npts, int C)
{
    size_t total = (size_t)B_ * Npts * C;
    size_t i = blockIdx.x * (size_t)blockDim.x + threadIdx.x;
    if (i >= total) return;
    size_t per = (size_t)Npts * C;
    int b = (int)(i / per);
    size_t rem = i % per;
    int c = (int)(rem / Npts);
    int n = (int)(rem % Npts);
    out[i] = in[((size_t)b * Npts + n) * C + c];
}

// ============================ Host driver ==================================
static inline dim3 grid1d(size_t n, int t) { return dim3((unsigned)((n + t - 1) / t)); }

extern "C" void kernel_launch(void* const* d_in, const int* in_sizes, int n_in,
                              void* d_out, int out_size, void* d_ws, size_t ws_size,
                              hipStream_t stream)
{
    (void)in_sizes; (void)n_in; (void)out_size; (void)ws_size;
    auto F = [&](int i) { return (const float*)d_in[i]; };

    const float* P = F(0);  // (B, N, 4): xyz stride 4, feat = P+3 stride 4

    // fp[0]: dims [257,128,128]
    const float *fp0_b0 = F(1),  *fp0_s0 = F(2),  *fp0_w0 = F(3);
    const float *fp0_b1 = F(4),  *fp0_s1 = F(5),  *fp0_w1 = F(6);
    // fp[1]: dims [352,256,256]
    const float *fp1_b0 = F(7),  *fp1_s0 = F(8),  *fp1_w0 = F(9);
    const float *fp1_b1 = F(10), *fp1_s1 = F(11), *fp1_w1 = F(12);
    // sa[0][0]: [4,16,16,32]
    const float *s00_b0 = F(13), *s00_s0 = F(14), *s00_w0 = F(15);
    const float *s00_b1 = F(16), *s00_s1 = F(17), *s00_w1 = F(18);
    const float *s00_b2 = F(19), *s00_s2 = F(20), *s00_w2 = F(21);
    // sa[0][1]: [4,32,32,64]
    const float *s01_b0 = F(22), *s01_s0 = F(23), *s01_w0 = F(24);
    const float *s01_b1 = F(25), *s01_s1 = F(26), *s01_w1 = F(27);
    const float *s01_b2 = F(28), *s01_s2 = F(29), *s01_w2 = F(30);
    // sa[1][0]: [99,64,64,128]
    const float *s10_b0 = F(31), *s10_s0 = F(32), *s10_w0 = F(33);
    const float *s10_b1 = F(34), *s10_s1 = F(35), *s10_w1 = F(36);
    const float *s10_b2 = F(37), *s10_s2 = F(38), *s10_w2 = F(39);
    // sa[1][1]: [99,64,96,128]
    const float *s11_b0 = F(40), *s11_s0 = F(41), *s11_w0 = F(42);
    const float *s11_b1 = F(43), *s11_s1 = F(44), *s11_w1 = F(45);
    const float *s11_b2 = F(46), *s11_s2 = F(47), *s11_w2 = F(48);

    // -------- workspace carve-up (~100 MB) --------
    float* ws = (float*)d_ws;
    size_t off = 0;
    auto alloc = [&](size_t n) { float* p = ws + off; off += n; return p; };
    float* fps_dist = alloc((size_t)BATCH * NPTS);
    float* new_xyz1 = alloc((size_t)BATCH * 1024 * 3);
    float* new_xyz2 = alloc((size_t)BATCH * 256 * 3);
    float* grp10    = alloc((size_t)BATCH * 1024 * 16 * 4);
    float* grp11    = alloc((size_t)BATCH * 1024 * 32 * 4);
    float* grp20    = alloc((size_t)BATCH * 256 * 16 * 99);
    float* grp21    = alloc((size_t)BATCH * 256 * 32 * 99);
    float* feat1    = alloc((size_t)BATCH * 1024 * 96);
    float* feat2    = alloc((size_t)BATCH * 256 * 256);
    float* newfeat1 = alloc((size_t)BATCH * 1024 * 256);
    float* w3_1     = alloc((size_t)BATCH * 1024 * 3);
    float* w3_0     = alloc((size_t)BATCH * NPTS * 3);
    float* bufA     = alloc((size_t)8388608);
    float* bufB     = alloc((size_t)8388608);
    int* ip = (int*)(ws + off);
    int* fps_idx1 = ip; ip += BATCH * 1024;
    int* fps_idx2 = ip; ip += BATCH * 256;
    int* idx3_1   = ip; ip += BATCH * 1024 * 3;
    int* idx3_0   = ip; ip += BATCH * NPTS * 3;

    float* out = (float*)d_out;
    const size_t OUT1_OFF = (size_t)BATCH * 128 * NPTS;             // 8388608
    const size_t OUT2_OFF = OUT1_OFF + (size_t)BATCH * 256 * 1024;  // 9437184

    auto gemm = [&](const float* A, const float* W, const float* sc,
                    const float* bi, float* o, int M, int K, int N,
                    const int* id3, const float* w3p, const float* kf,
                    int Ki, const float* sf, int ss) {
        dim3 g((N + TN - 1) / TN, (M + TM - 1) / TM);
        wmma_gemm_bias_relu<<<g, 256, 0, stream>>>(A, W, sc, bi, o, M, K, N,
                                                   id3, w3p, kf, Ki, sf, ss);
    };

    // ======================= SA level 1 =======================
    fps_kernel<<<BATCH, 256, 0, stream>>>(P, 4, NPTS, 1024, fps_dist, fps_idx1);
    gather_xyz_kernel<<<grid1d((size_t)BATCH * 1024, 256), 256, 0, stream>>>(
        P, 4, fps_idx1, BATCH, NPTS, 1024, new_xyz1);
    ball_group_kernel<<<grid1d((size_t)BATCH * 1024, 256), 256, 0, stream>>>(
        P, 4, P + 3, 1, 4, new_xyz1, BATCH, NPTS, 1024, 16, 0.01f, grp10);
    ball_group_kernel<<<grid1d((size_t)BATCH * 1024, 256), 256, 0, stream>>>(
        P, 4, P + 3, 1, 4, new_xyz1, BATCH, NPTS, 1024, 32, 0.25f, grp11);

    // scale 0: [4 ->16 ->16 ->32], M = 4*1024*16 = 65536
    gemm(grp10, s00_w0, s00_s0, s00_b0, bufA, 65536, 4, 16, nullptr, nullptr, nullptr, 0, nullptr, 0);
    gemm(bufA,  s00_w1, s00_s1, s00_b1, bufB, 65536, 16, 16, nullptr, nullptr, nullptr, 0, nullptr, 0);
    gemm(bufB,  s00_w2, s00_s2, s00_b2, bufA, 65536, 16, 32, nullptr, nullptr, nullptr, 0, nullptr, 0);
    maxpool_kernel<<<grid1d((size_t)BATCH * 1024 * 32, 256), 256, 0, stream>>>(
        bufA, 16, 32, feat1, 96, 0, (size_t)BATCH * 1024 * 32);

    // scale 1: [4 ->32 ->32 ->64], M = 4*1024*32 = 131072
    gemm(grp11, s01_w0, s01_s0, s01_b0, bufA, 131072, 4, 32, nullptr, nullptr, nullptr, 0, nullptr, 0);
    gemm(bufA,  s01_w1, s01_s1, s01_b1, bufB, 131072, 32, 32, nullptr, nullptr, nullptr, 0, nullptr, 0);
    gemm(bufB,  s01_w2, s01_s2, s01_b2, bufA, 131072, 32, 64, nullptr, nullptr, nullptr, 0, nullptr, 0);
    maxpool_kernel<<<grid1d((size_t)BATCH * 1024 * 64, 256), 256, 0, stream>>>(
        bufA, 32, 64, feat1, 96, 32, (size_t)BATCH * 1024 * 64);

    // ======================= SA level 2 =======================
    fps_kernel<<<BATCH, 256, 0, stream>>>(new_xyz1, 3, 1024, 256, fps_dist, fps_idx2);
    gather_xyz_kernel<<<grid1d((size_t)BATCH * 256, 256), 256, 0, stream>>>(
        new_xyz1, 3, fps_idx2, BATCH, 1024, 256, new_xyz2);
    ball_group_kernel<<<grid1d((size_t)BATCH * 256, 256), 256, 0, stream>>>(
        new_xyz1, 3, feat1, 96, 96, new_xyz2, BATCH, 1024, 256, 16, 0.25f, grp20);
    ball_group_kernel<<<grid1d((size_t)BATCH * 256, 256), 256, 0, stream>>>(
        new_xyz1, 3, feat1, 96, 96, new_xyz2, BATCH, 1024, 256, 32, 1.0f, grp21);

    // scale 0: [99 ->64 ->64 ->128], M = 4*256*16 = 16384
    gemm(grp20, s10_w0, s10_s0, s10_b0, bufA, 16384, 99, 64, nullptr, nullptr, nullptr, 0, nullptr, 0);
    gemm(bufA,  s10_w1, s10_s1, s10_b1, bufB, 16384, 64, 64, nullptr, nullptr, nullptr, 0, nullptr, 0);
    gemm(bufB,  s10_w2, s10_s2, s10_b2, bufA, 16384, 64, 128, nullptr, nullptr, nullptr, 0, nullptr, 0);
    maxpool_kernel<<<grid1d((size_t)BATCH * 256 * 128, 256), 256, 0, stream>>>(
        bufA, 16, 128, feat2, 256, 0, (size_t)BATCH * 256 * 128);

    // scale 1: [99 ->64 ->96 ->128], M = 4*256*32 = 32768
    gemm(grp21, s11_w0, s11_s0, s11_b0, bufA, 32768, 99, 64, nullptr, nullptr, nullptr, 0, nullptr, 0);
    gemm(bufA,  s11_w1, s11_s1, s11_b1, bufB, 32768, 64, 96, nullptr, nullptr, nullptr, 0, nullptr, 0);
    gemm(bufB,  s11_w2, s11_s2, s11_b2, bufA, 32768, 96, 128, nullptr, nullptr, nullptr, 0, nullptr, 0);
    maxpool_kernel<<<grid1d((size_t)BATCH * 256 * 128, 256), 256, 0, stream>>>(
        bufA, 32, 128, feat2, 256, 128, (size_t)BATCH * 256 * 128);

    // ======================= FP stage 1 (level2 -> level1) =======================
    three_nn_kernel<<<grid1d((size_t)BATCH * 1024, 256), 256, 0, stream>>>(
        new_xyz1, 3, new_xyz2, 3, BATCH, 1024, 256, idx3_1, w3_1);
    // K = 256 (interp from feat2) + 96 (skip feat1) = 352
    gemm(nullptr, fp1_w0, fp1_s0, fp1_b0, bufA, BATCH * 1024, 352, 256,
         idx3_1, w3_1, feat2, 256, feat1, 96);
    gemm(bufA, fp1_w1, fp1_s1, fp1_b1, newfeat1, BATCH * 1024, 256, 256,
         nullptr, nullptr, nullptr, 0, nullptr, 0);

    // ======================= FP stage 0 (level1 -> level0) =======================
    three_nn_kernel<<<grid1d((size_t)BATCH * NPTS, 256), 256, 0, stream>>>(
        P, 4, new_xyz1, 3, BATCH, NPTS, 1024, idx3_0, w3_0);
    // K = 256 (interp from newfeat1) + 1 (skip = points channel 3, stride 4) = 257
    gemm(nullptr, fp0_w0, fp0_s0, fp0_b0, bufA, BATCH * NPTS, 257, 128,
         idx3_0, w3_0, newfeat1, 256, P + 3, 4);
    gemm(bufA, fp0_w1, fp0_s1, fp0_b1, bufB, BATCH * NPTS, 128, 128,
         nullptr, nullptr, nullptr, 0, nullptr, 0);

    // ======================= outputs (transpose to B,C,N) =======================
    transpose_kernel<<<grid1d((size_t)BATCH * NPTS * 128, 256), 256, 0, stream>>>(
        bufB, out, BATCH, NPTS, 128);
    transpose_kernel<<<grid1d((size_t)BATCH * 1024 * 256, 256), 256, 0, stream>>>(
        newfeat1, out + OUT1_OFF, BATCH, 1024, 256);
    transpose_kernel<<<grid1d((size_t)BATCH * 256 * 256, 256), 256, 0, stream>>>(
        feat2, out + OUT2_OFF, BATCH, 256, 256);
}
